// GraphAttentionLayer_29910152249389
// MI455X (gfx1250) — matrix operand order repaired and checked by reading, not compiled, not run on hardware
//
#include <hip/hip_runtime.h>

#define ALPHA_   0.2f
#define MASKV_  -9.0e15f
#define LOG2E_   1.4426950408889634f
#define B_       8
#define N_       2048
#define FIN_     64
#define FOUT_    64

typedef __attribute__((ext_vector_type(2)))  float    v2f;
typedef __attribute__((ext_vector_type(4)))  float    v4f;
typedef __attribute__((ext_vector_type(8)))  float    v8f;
typedef __attribute__((ext_vector_type(4)))  int      v4i;
typedef __attribute__((ext_vector_type(8)))  _Float16 v8h;
typedef __attribute__((ext_vector_type(16))) _Float16 v16h;

// ---------------------------------------------------------------------------
// Stage 1: h = x @ W^T (fp32 WMMA 16x16x4), then
//   e1 = (h @ a1) * log2(e), e2 = (h @ a2) * log2(e)   [exp2-domain scores:
//   leaky_relu commutes with positive scaling, so stage 2 can use raw
//   v_exp_f32 with no per-element multiply]
//   ht16[b][o][n] = (f16) h[b][n][o]   (transposed store for stage-2 B-operand)
// One wave handles one (batch, 16-row tile). 1024 waves total.
// ---------------------------------------------------------------------------
__global__ __launch_bounds__(256) void gat_stage1(
    const float* __restrict__ x, const float* __restrict__ W,
    const float* __restrict__ a, _Float16* __restrict__ ht16,
    float* __restrict__ e1, float* __restrict__ e2)
{
    const int wave = (blockIdx.x * blockDim.x + threadIdx.x) >> 5;
    const int lane = threadIdx.x & 31;
    const int b    = wave >> 7;          // 128 row-tiles per batch
    const int i0   = (wave & 127) << 4;
    const int row  = lane & 15;
    const int hi   = lane >> 4;

    // A operand: x[b][i0+row][k0 + 2*hi + {0,1}]  (16x4 f32 A layout)
    const float* xrow = x + ((size_t)b * N_ + (i0 + row)) * FIN_;

    v8f c[4];
    #pragma unroll
    for (int t = 0; t < 4; ++t) c[t] = (v8f){};

    #pragma unroll
    for (int k0 = 0; k0 < FIN_; k0 += 4) {
        v2f av = *(const v2f*)(xrow + k0 + 2 * hi);
        #pragma unroll
        for (int t = 0; t < 4; ++t) {
            // B operand: W[o][k0 + 2*hi + {0,1}], o = t*16 + row (4x16 f32 B layout)
            v2f bv = *(const v2f*)(W + (size_t)(t * 16 + row) * FIN_ + k0 + 2 * hi);
            c[t] = __builtin_amdgcn_wmma_f32_16x16x4_f32(
                false, av, false, bv, (short)0, c[t], false, false);
        }
    }
    // C layout: c[t][v] = h[b][i0 + v + 8*hi][t*16 + row]

    // ---- e1 / e2: per-lane partials, reduce over the 16 o-lanes of each half
    float a1v[4], a2v[4];
    #pragma unroll
    for (int t = 0; t < 4; ++t) {
        a1v[t] = a[t * 16 + row];
        a2v[t] = a[FOUT_ + t * 16 + row];
    }
    float p1[8], p2[8];
    #pragma unroll
    for (int v = 0; v < 8; ++v) {
        p1[v] = c[0][v]*a1v[0] + c[1][v]*a1v[1] + c[2][v]*a1v[2] + c[3][v]*a1v[3];
        p2[v] = c[0][v]*a2v[0] + c[1][v]*a2v[1] + c[2][v]*a2v[2] + c[3][v]*a2v[3];
    }
    #pragma unroll
    for (int m = 1; m < 16; m <<= 1) {
        #pragma unroll
        for (int v = 0; v < 8; ++v) {
            p1[v] += __shfl_xor(p1[v], m);
            p2[v] += __shfl_xor(p2[v], m);
        }
    }
    if (row == 0) {
        #pragma unroll
        for (int v = 0; v < 8; ++v) {
            e1[(size_t)b * N_ + i0 + v + 8 * hi] = p1[v] * LOG2E_;
            e2[(size_t)b * N_ + i0 + v + 8 * hi] = p2[v] * LOG2E_;
        }
    }

    // ---- transposed f16 store: ht16[b][o][n], n contiguous over v
    #pragma unroll
    for (int t = 0; t < 4; ++t) {
        v8h hp;
        #pragma unroll
        for (int v = 0; v < 8; ++v) hp[v] = (_Float16)c[t][v];
        *(v8h*)(ht16 + ((size_t)b * FOUT_ + t * 16 + row) * N_ + i0 + 8 * hi) = hp;
    }
}

// ---------------------------------------------------------------------------
// Per-batch max of e2 -> m2[b].  Gives an exact upper bound on every score
// row:  s_ij = leaky(e1_i + e2_j) <= leaky(e1_i + m2_b)  (leaky is monotone;
// everything already in exp2 domain, which preserves ordering since c>0).
// ---------------------------------------------------------------------------
__global__ __launch_bounds__(256) void gat_e2max(
    const float* __restrict__ e2, float* __restrict__ m2)
{
    const int b   = blockIdx.x;
    const int tid = threadIdx.x;
    const float* p = e2 + (size_t)b * N_;
    float m = -__builtin_inff();
    for (int j = tid; j < N_; j += 256) m = fmaxf(m, p[j]);
    #pragma unroll
    for (int s = 1; s < 32; s <<= 1) m = fmaxf(m, __shfl_xor(m, s));
    __shared__ float sm[8];
    if ((tid & 31) == 0) sm[tid >> 5] = m;
    __syncthreads();
    if (tid == 0) {
        float mm = sm[0];
        #pragma unroll
        for (int i = 1; i < 8; ++i) mm = fmaxf(mm, sm[i]);
        m2[b] = mm;
    }
}

// ---------------------------------------------------------------------------
// Stage 2: fused masked-softmax attention + P @ H, single pass (fixed row
// bound m_i, so p in [0,1] fits f16 and partial sums are addable).
// Block = 512 threads = 16 waves = 8 batches x 2 j-halves, all sharing one
// 16-row i-tile -> adj reads hit the WGP cache 8x. Halves combine via LDS.
// p = 2^(score - bound) via raw v_exp_f32 (scores pre-scaled by log2e).
// ---------------------------------------------------------------------------
__global__ __launch_bounds__(512) void gat_stage2(
    const int* __restrict__ adj, const _Float16* __restrict__ ht16,
    const float* __restrict__ e1, const float* __restrict__ e2,
    const float* __restrict__ m2, float* __restrict__ out)
{
    __shared__ float lacc[B_][32 * 32];   // partner-half accumulators
    __shared__ float lsmb[B_][32];        // partner-half row sums

    const int tid  = threadIdx.x;
    const int lane = tid & 31;
    const int w    = tid >> 5;            // 0..15
    const int b    = w & 7;
    const int half = w >> 3;
    const int i0   = blockIdx.x << 4;
    const int row  = lane & 15;
    const int hi   = lane >> 4;

    const float ei = e1[(size_t)b * N_ + i0 + row];   // already * log2e
    float m_row = ei + m2[b];
    m_row = fmaxf(m_row, m_row * ALPHA_);             // leaky(upper bound)

    const int*      adjrow = adj + (size_t)(i0 + row) * N_;
    const float*    e2b    = e2 + (size_t)b * N_;
    const _Float16* hb     = ht16 + (size_t)b * FOUT_ * N_;

    v8f acc[4];
    #pragma unroll
    for (int t = 0; t < 4; ++t) acc[t] = (v8f){};
    float l_run = 0.f;

    const int jbeg = half * (N_ / 2);
    const int jend = jbeg + (N_ / 2);

    #pragma unroll 2
    for (int j0 = jbeg; j0 < jend; j0 += 32) {
        if (j0 + 32 < jend) {
            __builtin_prefetch(adjrow + j0 + 32, 0, 0);   // global_prefetch_b8
        }
        // ---- P = 2^(masked leaky score - m_row) in 16x32 f16 A lane layout:
        // half-slot hv = g*4+cc maps to K = (g&1)*4 + (g>>1)*16 + 8*hi + cc
        v16h pA;
        float lsum = 0.f;
        #pragma unroll
        for (int g = 0; g < 4; ++g) {
            const int kb = ((g & 1) << 2) + ((g >> 1) << 4) + (hi << 3);
            v4i av = *(const v4i*)(adjrow + j0 + kb);
            v4f ev = *(const v4f*)(e2b + j0 + kb);
            #pragma unroll
            for (int cc = 0; cc < 4; ++cc) {
                float sv = ei + ev[cc];
                sv = fmaxf(sv, sv * ALPHA_);              // leaky_relu (exp2 dom.)
                sv = (av[cc] > 0) ? sv : MASKV_;
                float p = __builtin_amdgcn_exp2f(sv - m_row);  // in [0,1]; masked -> 0
                lsum += p;
                pA[g * 4 + cc] = (_Float16)p;
            }
        }
        l_run += lsum;

        // ---- acc[t] += P(16x32) @ H(32x16) for each o-tile t
        #pragma unroll
        for (int t = 0; t < 4; ++t) {
            // B layout: lane n = row -> o = t*16+row; hv<8 -> K = 8*hi + hv, hv>=8 -> K = 16 + 8*hi + (hv-8)
            const _Float16* src = hb + (size_t)(t * 16 + row) * N_ + j0 + (hi << 3);
            v8h lo  = *(const v8h*)(src);
            v8h hi8 = *(const v8h*)(src + 16);
            v16h pB = __builtin_shufflevector(lo, hi8,
                0, 1, 2, 3, 4, 5, 6, 7, 8, 9, 10, 11, 12, 13, 14, 15);
            acc[t] = __builtin_amdgcn_wmma_f32_16x16x32_f16(
                false, pA, false, pB, (short)0, acc[t], false, false);
        }
    }

    // ---- combine the two j-halves through LDS (same fixed m_row -> plain add)
    if (half == 1) {
        #pragma unroll
        for (int t = 0; t < 4; ++t)
            #pragma unroll
            for (int v = 0; v < 8; ++v)
                lacc[b][(t * 8 + v) * 32 + lane] = acc[t][v];
        lsmb[b][lane] = l_run;
    }
    __syncthreads();
    if (half == 1) return;

    #pragma unroll
    for (int t = 0; t < 4; ++t)
        #pragma unroll
        for (int v = 0; v < 8; ++v)
            acc[t][v] += lacc[b][(t * 8 + v) * 32 + lane];
    l_run += lsmb[b][lane];

    // ---- finalize: divide by row sum (combined across halves), ReLU, store
    l_run += __shfl_xor(l_run, 16);
    const float inv = 1.f / l_run;
    float* ob = out + ((size_t)b * N_ + i0) * FOUT_;
    #pragma unroll
    for (int v = 0; v < 8; ++v) {
        float iv = __shfl(inv, v + (hi << 3));
        const int r = v + 8 * hi;
        #pragma unroll
        for (int t = 0; t < 4; ++t) {
            float val = acc[t][v] * iv;
            ob[(size_t)r * FOUT_ + t * 16 + row] = fmaxf(val, 0.f);
        }
    }
}

// ---------------------------------------------------------------------------
extern "C" void kernel_launch(void* const* d_in, const int* in_sizes, int n_in,
                              void* d_out, int out_size, void* d_ws, size_t ws_size,
                              hipStream_t stream) {
    const float* x   = (const float*)d_in[0];   // (8, 2048, 64) f32
    const int*   adj = (const int*)d_in[1];     // (2048, 2048) i32
    const float* W   = (const float*)d_in[2];   // (64, 64) f32
    const float* a   = (const float*)d_in[3];   // (128, 1) f32
    float*       out = (float*)d_out;           // (8, 2048, 64) f32

    char* ws = (char*)d_ws;
    _Float16* ht16 = (_Float16*)ws;                                   // 2 MB
    float*    e1   = (float*)(ws + (size_t)B_ * FOUT_ * N_ * sizeof(_Float16));
    float*    e2   = e1 + (size_t)B_ * N_;                            // 64 KB each
    float*    m2   = e2 + (size_t)B_ * N_;                            // 8 floats

    gat_stage1<<<dim3(128), dim3(256), 0, stream>>>(x, W, a, ht16, e1, e2);
    gat_e2max <<<dim3(B_),  dim3(256), 0, stream>>>(e2, m2);
    // one block per i-tile: 8 batches x 2 j-halves = 16 waves
    gat_stage2<<<dim3(128), dim3(512), 0, stream>>>(adj, ht16, e1, e2, m2, out);
}
